// AttentionLayer_75565654606299
// MI455X (gfx1250) — matrix operand order repaired and checked by reading, not compiled, not run on hardware
//
#include <hip/hip_runtime.h>
#include <hip/hip_bf16.h>
#include <cstdint>

// Problem sizes (fixed by the reference)
#define N_TOK 4096
#define D_DIM 1024

// Tiling
#define BM 128
#define BN 128
#define BK 32
#define NTHR 256           // 8 waves (wave32)
#define LDS_STRIDE 40      // BK + 8 pad; 80B row stride keeps 16B alignment

typedef __attribute__((ext_vector_type(16))) __bf16 bf16x16;
typedef __attribute__((ext_vector_type(8)))  __bf16 bf16x8;
typedef __attribute__((ext_vector_type(8)))  float  f32x8;
typedef __attribute__((ext_vector_type(4)))  int    i32x4;
typedef __attribute__((ext_vector_type(8)))  int    i32x8;
typedef __attribute__((ext_vector_type(4)))  unsigned int u32x4;

// ---------------------------------------------------------------------------
// CDNA5 async global->LDS copy (ASYNCcnt path).
// ---------------------------------------------------------------------------
#if defined(__has_builtin)
#  if __has_builtin(__builtin_amdgcn_global_load_async_to_lds_b128)
#    define HAVE_ASYNC_LDS 1
#  endif
#  if __has_builtin(__builtin_amdgcn_tensor_load_to_lds) && \
      __has_builtin(__builtin_amdgcn_s_wait_tensorcnt)
#    define HAVE_TDM 1
#  endif
#endif
#ifndef HAVE_ASYNC_LDS
#  define HAVE_ASYNC_LDS 0
#endif
#ifndef HAVE_TDM
#  define HAVE_TDM 0
#endif

__device__ __forceinline__ void async_copy16(const __bf16* g, __bf16* l) {
#if HAVE_ASYNC_LDS
  __builtin_amdgcn_global_load_async_to_lds_b128(
      (__attribute__((address_space(1))) i32x4*)g,
      (__attribute__((address_space(3))) i32x4*)l, 0, 0);
#else
  *(bf16x8*)l = *(const bf16x8*)g;
#endif
}

__device__ __forceinline__ void wait_async() {
#if HAVE_ASYNC_LDS
#  if defined(__has_builtin) && __has_builtin(__builtin_amdgcn_s_wait_asynccnt)
  __builtin_amdgcn_s_wait_asynccnt(0);
#  else
  asm volatile("s_wait_asynccnt 0" ::: "memory");
#  endif
#endif
}

// ---------------------------------------------------------------------------
// Tensor Data Mover: load a BM x BK bf16 tile (row-major, leading dim
// ld_elems) into LDS with hardware padding of 4 DWORDs after every 16 DWORDs
// -> reproduces the LDS_STRIDE=40-element row layout.  Issue from ONE wave.
// This toolchain's builtin takes 6 args: (g0, g1, g2, g3, g4, cpol).
// ---------------------------------------------------------------------------
__device__ __forceinline__ unsigned int lds_addr_u32(const void* p) {
  return (unsigned int)(uintptr_t)(__attribute__((address_space(3))) const void*)p;
}

__device__ __forceinline__ void tdm_load_tile(const __bf16* gsrc,
                                              unsigned int lds_byte_addr,
                                              int ld_elems, int n_rows) {
#if HAVE_TDM
  const unsigned long long ga = (unsigned long long)(uintptr_t)gsrc;
  u32x4 g0;
  g0[0] = 1u;                                        // count=1, user D#
  g0[1] = lds_byte_addr;                             // LDS dest (bytes)
  g0[2] = (unsigned int)(ga & 0xFFFFFFFFu);          // global_addr[31:0]
  g0[3] = (unsigned int)((ga >> 32) & 0x01FFFFFFu)   // global_addr[56:32]
        | (2u << 30);                                // type=2 (image)
  i32x8 g1;
  g1[0] = (1 << 16)            // data_size = 1 -> 2 bytes
        | (1 << 20)            // pad_enable
        | (3 << 22)            // pad_interval code 3 -> every 16 DWORDs
        | (3 << 25);           // pad_amount  code 3 -> 4 DWORDs
  g1[1] = (ld_elems & 0xFFFF) << 16;                        // tensor_dim0[15:0]
  g1[2] = ((ld_elems >> 16) & 0xFFFF) | ((n_rows & 0xFFFF) << 16);
  g1[3] = ((n_rows >> 16) & 0xFFFF) | (BK << 16);           // tile_dim0 = 32
  g1[4] = BM;                                               // tile_dim1 = 128
  g1[5] = ld_elems;                                         // dim0_stride[31:0]
  g1[6] = 0;
  g1[7] = 0;
  i32x4 z4 = {0, 0, 0, 0};
  i32x8 z8 = {0, 0, 0, 0, 0, 0, 0, 0};
  __builtin_amdgcn_tensor_load_to_lds(g0, g1, z4, z4, z8, 0);
#else
  (void)gsrc; (void)lds_byte_addr; (void)ld_elems; (void)n_rows;
#endif
}

__device__ __forceinline__ void wait_tensor() {
#if HAVE_TDM
  __builtin_amdgcn_s_wait_tensorcnt(0);
#endif
}

// ---------------------------------------------------------------------------
// LDS tile fills.  Layout for BOTH operands is tile[row][k] (row = M or N).
// ---------------------------------------------------------------------------
__device__ __forceinline__ void fill_a_f32(__bf16 (*As)[LDS_STRIDE],
                                           const float* __restrict__ A,
                                           int row0, int k0, int ldA, int tid) {
  float4 f[4];
#pragma unroll
  for (int i = 0; i < 4; ++i) {
    int id = tid + i * NTHR;                 // 0..1023
    int r  = id >> 3;                        // 8 float4 per row
    int c  = (id & 7) << 2;
    f[i] = *(const float4*)(A + (size_t)(row0 + r) * ldA + k0 + c);
  }
#pragma unroll
  for (int i = 0; i < 4; ++i) {
    int id = tid + i * NTHR;
    int r  = id >> 3;
    int c  = (id & 7) << 2;
    As[r][c + 0] = (__bf16)f[i].x;
    As[r][c + 1] = (__bf16)f[i].y;
    As[r][c + 2] = (__bf16)f[i].z;
    As[r][c + 3] = (__bf16)f[i].w;
  }
}

__device__ __forceinline__ void fill_b_f32_nt(__bf16 (*Bs)[LDS_STRIDE],
                                              const float* __restrict__ B,
                                              int k0, int col0, int ldB, int tid) {
  float4 f[4];
#pragma unroll
  for (int i = 0; i < 4; ++i) {
    int id = tid + i * NTHR;                 // 0..1023
    int k  = id >> 5;                        // 32 float4 per k-row
    int n  = (id & 31) << 2;
    f[i] = *(const float4*)(B + (size_t)(k0 + k) * ldB + col0 + n);
  }
#pragma unroll
  for (int i = 0; i < 4; ++i) {
    int id = tid + i * NTHR;
    int k  = id >> 5;
    int n  = (id & 31) << 2;
    Bs[n + 0][k] = (__bf16)f[i].x;
    Bs[n + 1][k] = (__bf16)f[i].y;
    Bs[n + 2][k] = (__bf16)f[i].z;
    Bs[n + 3][k] = (__bf16)f[i].w;
  }
}

// tile[r][k] = M_bf16[row0+r][k0+k] (source row-major along k): async copy.
__device__ __forceinline__ void fill_bf16_async(__bf16 (*T)[LDS_STRIDE],
                                                const __bf16* __restrict__ A,
                                                int row0, int k0, int ldA, int tid) {
#pragma unroll
  for (int i = 0; i < 2; ++i) {
    int id = tid + i * NTHR;                 // 0..511
    int r  = id >> 2;                        // 4 chunks of 8 per row
    int c  = (id & 3) << 3;
    async_copy16(A + (size_t)(row0 + r) * ldA + k0 + c, &T[r][c]);
  }
}

// ---------------------------------------------------------------------------
// Fragment gathers (ISA 16-bit 16x32 A layout).
// ---------------------------------------------------------------------------
__device__ __forceinline__ bf16x16 frag_a(const __bf16* rowp, int half) {
  bf16x8 lo = *(const bf16x8*)(rowp + half * 8);
  bf16x8 hi = *(const bf16x8*)(rowp + 16 + half * 8);
  bf16x16 r;
#pragma unroll
  for (int i = 0; i < 8; ++i) { r[i] = lo[i]; r[i + 8] = hi[i]; }
  return r;
}
__device__ __forceinline__ bf16x16 frag_b(const __bf16* rowp, int half) {
  bf16x8 lo = *(const bf16x8*)(rowp + half * 8);
  bf16x8 hi = *(const bf16x8*)(rowp + 16 + half * 8);
  bf16x16 r;
#pragma unroll
  for (int i = 0; i < 8; ++i) { r[i] = lo[i]; r[i + 8] = hi[i]; }
  return r;
}

__device__ __forceinline__ f32x8 wmma_bf16(bf16x16 a, bf16x16 b, f32x8 c) {
  return __builtin_amdgcn_wmma_f32_16x16x32_bf16(false, a, false, b, (short)0, c,
                                                 false, false);
}

// Batch all 8 B-fragment loads before the 8 WMMAs so the DS counter can be
// counted down instead of a full wait per WMMA.
__device__ __forceinline__ void compute_step(const __bf16 (*As)[LDS_STRIDE],
                                             const __bf16 (*Bs)[LDS_STRIDE],
                                             f32x8 (&acc)[8],
                                             int wave, int ln, int half) {
  bf16x16 a = frag_a(&As[wave * 16 + ln][0], half);
  bf16x16 b[8];
#pragma unroll
  for (int j = 0; j < 8; ++j) b[j] = frag_b(&Bs[j * 16 + ln][0], half);
#pragma unroll
  for (int j = 0; j < 8; ++j) acc[j] = wmma_bf16(a, b[j], acc[j]);
}

// ---------------------------------------------------------------------------
// Kernel 1: projection GEMM  out = x @ W + b   (f32 inputs, bf16 or f32 out)
// ---------------------------------------------------------------------------
__global__ __launch_bounds__(NTHR) void proj_gemm_kernel(
    const float* __restrict__ X, const float* __restrict__ W,
    const float* __restrict__ bias, __bf16* __restrict__ outBF,
    float* __restrict__ outF, int storeBF) {
  __shared__ __attribute__((aligned(16))) __bf16 As[2][BM][LDS_STRIDE];
  __shared__ __attribute__((aligned(16))) __bf16 Bs[2][BN][LDS_STRIDE];
  const int tid  = threadIdx.x;
  const int lane = tid & 31, wave = tid >> 5;
  const int half = lane >> 4, ln = lane & 15;
  const int row0 = blockIdx.y * BM, col0 = blockIdx.x * BN;

  f32x8 acc[8];
#pragma unroll
  for (int j = 0; j < 8; ++j)
#pragma unroll
    for (int r = 0; r < 8; ++r) acc[j][r] = 0.f;

  fill_a_f32(As[0], X, row0, 0, D_DIM, tid);
  fill_b_f32_nt(Bs[0], W, 0, col0, D_DIM, tid);
  __syncthreads();

  for (int k0 = 0; k0 < D_DIM; k0 += BK) {
    const int cur = (k0 >> 5) & 1;
    if (k0 + BK < D_DIM) {
      __builtin_prefetch(X + (size_t)(row0 + (tid >> 1)) * D_DIM + k0 + BK, 0, 1);
      fill_a_f32(As[cur ^ 1], X, row0, k0 + BK, D_DIM, tid);
      fill_b_f32_nt(Bs[cur ^ 1], W, k0 + BK, col0, D_DIM, tid);
    }
    compute_step(As[cur], Bs[cur], acc, wave, ln, half);
    __syncthreads();
  }

#pragma unroll
  for (int j = 0; j < 8; ++j) {
    const int col = col0 + j * 16 + ln;
    const float bv = bias[col];
#pragma unroll
    for (int r = 0; r < 8; ++r) {
      const int row = row0 + wave * 16 + half * 8 + r;
      const float v = acc[j][r] + bv;
      if (storeBF) outBF[(size_t)row * D_DIM + col] = (__bf16)v;
      else         outF [(size_t)row * D_DIM + col] = v;
    }
  }
}

// ---------------------------------------------------------------------------
// Kernel 2: scores  E = exp(mask * (Q @ K^T)), rowsum[i] += sum_j E[i][j]
// Async (ASYNCcnt) double-buffered fills.
// ---------------------------------------------------------------------------
__global__ __launch_bounds__(NTHR) void scores_kernel(
    const __bf16* __restrict__ Q, const __bf16* __restrict__ Kmat,
    const float* __restrict__ mask, __bf16* __restrict__ E,
    float* __restrict__ rowsum) {
  __shared__ __attribute__((aligned(16))) __bf16 As[2][BM][LDS_STRIDE];
  __shared__ __attribute__((aligned(16))) __bf16 Bs[2][BN][LDS_STRIDE];
  const int tid  = threadIdx.x;
  const int lane = tid & 31, wave = tid >> 5;
  const int half = lane >> 4, ln = lane & 15;
  const int row0 = blockIdx.y * BM, col0 = blockIdx.x * BN;

  f32x8 acc[8];
#pragma unroll
  for (int j = 0; j < 8; ++j)
#pragma unroll
    for (int r = 0; r < 8; ++r) acc[j][r] = 0.f;

  fill_bf16_async(As[0], Q, row0, 0, D_DIM, tid);
  fill_bf16_async(Bs[0], Kmat, col0, 0, D_DIM, tid);
  wait_async();
  __syncthreads();

  for (int k0 = 0; k0 < D_DIM; k0 += BK) {
    const int cur = (k0 >> 5) & 1;
    if (k0 + BK < D_DIM) {
      fill_bf16_async(As[cur ^ 1], Q, row0, k0 + BK, D_DIM, tid);
      fill_bf16_async(Bs[cur ^ 1], Kmat, col0, k0 + BK, D_DIM, tid);
    }
    compute_step(As[cur], Bs[cur], acc, wave, ln, half);
    wait_async();
    __syncthreads();
  }

  float rsum[8];
#pragma unroll
  for (int r = 0; r < 8; ++r) rsum[r] = 0.f;

#pragma unroll
  for (int j = 0; j < 8; ++j) {
    const int col = col0 + j * 16 + ln;
#pragma unroll
    for (int r = 0; r < 8; ++r) {
      const int row = row0 + wave * 16 + half * 8 + r;
      const float s = acc[j][r] * mask[(size_t)row * N_TOK + col];
      const float e = __expf(s);
      E[(size_t)row * N_TOK + col] = (__bf16)e;
      rsum[r] += e;
    }
  }
#pragma unroll
  for (int r = 0; r < 8; ++r) {
    float v = rsum[r];
    v += __shfl_xor(v, 1);
    v += __shfl_xor(v, 2);
    v += __shfl_xor(v, 4);
    v += __shfl_xor(v, 8);
    if (ln == 0) {
      const int row = row0 + wave * 16 + half * 8 + r;
      atomicAdd(rowsum + row, v);
    }
  }
}

// ---------------------------------------------------------------------------
// Kernel 3: V'^T[d][n] = bf16(V[n][d] / rowsum[n])  via LDS 32x32 transpose.
// ---------------------------------------------------------------------------
__global__ __launch_bounds__(NTHR) void scale_v_kernel(
    const float* __restrict__ Vf, const float* __restrict__ rowsum,
    __bf16* __restrict__ VpT) {
  __shared__ __bf16 tile[32][40];
  const int r0 = blockIdx.x * 32;           // token rows
  const int c0 = blockIdx.y * 32;           // feature cols
  const int t  = threadIdx.x;
  const int lr = t >> 3;                    // 0..31
  const int lc = (t & 7) << 2;              // 0,4,..,28
  const float4 f = *(const float4*)(Vf + (size_t)(r0 + lr) * D_DIM + c0 + lc);
  const float inv = 1.0f / rowsum[r0 + lr];
  tile[lr][lc + 0] = (__bf16)(f.x * inv);
  tile[lr][lc + 1] = (__bf16)(f.y * inv);
  tile[lr][lc + 2] = (__bf16)(f.z * inv);
  tile[lr][lc + 3] = (__bf16)(f.w * inv);
  __syncthreads();
  const int oc = t >> 3;                    // feature within tile
  const int orr = (t & 7) << 2;             // token group
  __bf16* dst = VpT + (size_t)(c0 + oc) * N_TOK + r0 + orr;
#pragma unroll
  for (int e = 0; e < 4; ++e) dst[e] = tile[orr + e][oc];
}

// ---------------------------------------------------------------------------
// Kernel 4: out = E @ V'  (bf16 x bf16 -> f32).  Tiles DMA'd by the TDM with
// hardware LDS padding; wave 0 drives the descriptors, all waves sync on the
// barrier after wave 0's s_wait_tensorcnt.
// ---------------------------------------------------------------------------
__global__ __launch_bounds__(NTHR) void out_gemm_kernel(
    const __bf16* __restrict__ E, const __bf16* __restrict__ VpT,
    float* __restrict__ out) {
  __shared__ __attribute__((aligned(16))) __bf16 As[2][BM][LDS_STRIDE];
  __shared__ __attribute__((aligned(16))) __bf16 Bs[2][BN][LDS_STRIDE];
  const int tid  = threadIdx.x;
  const int lane = tid & 31, wave = tid >> 5;
  const int half = lane >> 4, ln = lane & 15;
  const int row0 = blockIdx.y * BM, col0 = blockIdx.x * BN;

  f32x8 acc[8];
#pragma unroll
  for (int j = 0; j < 8; ++j)
#pragma unroll
    for (int r = 0; r < 8; ++r) acc[j][r] = 0.f;

#if HAVE_TDM
  const unsigned int aBase = lds_addr_u32(&As[0][0][0]);
  const unsigned int bBase = lds_addr_u32(&Bs[0][0][0]);
  const unsigned int bufBytes = BM * LDS_STRIDE * 2;   // 10240 per buffer
  if (wave == 0) {
    tdm_load_tile(E   + (size_t)row0 * N_TOK, aBase, N_TOK, N_TOK);
    tdm_load_tile(VpT + (size_t)col0 * N_TOK, bBase, N_TOK, D_DIM);
    wait_tensor();
  }
#else
  fill_bf16_async(As[0], E, row0, 0, N_TOK, tid);
  fill_bf16_async(Bs[0], VpT, col0, 0, N_TOK, tid);
  wait_async();
#endif
  __syncthreads();

  for (int k0 = 0; k0 < N_TOK; k0 += BK) {
    const int cur = (k0 >> 5) & 1;
#if HAVE_TDM
    if (wave == 0 && k0 + BK < N_TOK) {
      tdm_load_tile(E   + (size_t)row0 * N_TOK + k0 + BK,
                    aBase + (unsigned)(cur ^ 1) * bufBytes, N_TOK, N_TOK);
      tdm_load_tile(VpT + (size_t)col0 * N_TOK + k0 + BK,
                    bBase + (unsigned)(cur ^ 1) * bufBytes, N_TOK, D_DIM);
    }
#else
    if (k0 + BK < N_TOK) {
      fill_bf16_async(As[cur ^ 1], E, row0, k0 + BK, N_TOK, tid);
      fill_bf16_async(Bs[cur ^ 1], VpT, col0, k0 + BK, N_TOK, tid);
    }
#endif
    compute_step(As[cur], Bs[cur], acc, wave, ln, half);
#if HAVE_TDM
    if (wave == 0) wait_tensor();
#else
    wait_async();
#endif
    __syncthreads();
  }

#pragma unroll
  for (int j = 0; j < 8; ++j) {
    const int col = col0 + j * 16 + ln;
#pragma unroll
    for (int r = 0; r < 8; ++r) {
      const int row = row0 + wave * 16 + half * 8 + r;
      out[(size_t)row * D_DIM + col] = acc[j][r];
    }
  }
}

// ---------------------------------------------------------------------------
__global__ void zero_f32_kernel(float* __restrict__ p, int n) {
  const int i = blockIdx.x * blockDim.x + threadIdx.x;
  if (i < n) p[i] = 0.f;
}

// ---------------------------------------------------------------------------
extern "C" void kernel_launch(void* const* d_in, const int* in_sizes, int n_in,
                              void* d_out, int out_size, void* d_ws, size_t ws_size,
                              hipStream_t stream) {
  (void)in_sizes; (void)n_in; (void)out_size; (void)ws_size;
  const float* x    = (const float*)d_in[0];
  const float* mask = (const float*)d_in[1];
  const float* Wq   = (const float*)d_in[2];
  const float* bq   = (const float*)d_in[3];
  const float* Wk   = (const float*)d_in[4];
  const float* bk   = (const float*)d_in[5];
  const float* Wv   = (const float*)d_in[6];
  const float* bv   = (const float*)d_in[7];
  float* out = (float*)d_out;

  char* ws = (char*)d_ws;
  const size_t MB = 1024 * 1024;
  __bf16* Qbf  = (__bf16*)(ws);             //  8 MB  [N, D] bf16
  __bf16* Kbf  = (__bf16*)(ws + 8 * MB);    //  8 MB  [N, D] bf16
  float*  Vf   = (float*) (ws + 16 * MB);   // 16 MB  [N, D] f32
  __bf16* VpT  = (__bf16*)(ws + 32 * MB);   //  8 MB  [D, N] bf16 (scaled, transposed)
  __bf16* Ebf  = (__bf16*)(ws + 40 * MB);   // 32 MB  [N, N] bf16
  float*  rsum = (float*) (ws + 72 * MB);   // 16 KB  [N]    f32

  const dim3 blk(NTHR);
  const dim3 gProj(D_DIM / BN, N_TOK / BM);   // (8, 32)
  const dim3 gScore(N_TOK / BN, N_TOK / BM);  // (32, 32)
  const dim3 gScale(N_TOK / 32, D_DIM / 32);  // (128, 32)

  proj_gemm_kernel<<<gProj, blk, 0, stream>>>(x, Wq, bq, Qbf, nullptr, 1);
  proj_gemm_kernel<<<gProj, blk, 0, stream>>>(x, Wk, bk, Kbf, nullptr, 1);
  proj_gemm_kernel<<<gProj, blk, 0, stream>>>(x, Wv, bv, nullptr, Vf, 0);
  zero_f32_kernel<<<N_TOK / 256, 256, 0, stream>>>(rsum, N_TOK);
  scores_kernel<<<gScore, blk, 0, stream>>>(Qbf, Kbf, mask, Ebf, rsum);
  scale_v_kernel<<<gScale, blk, 0, stream>>>(Vf, rsum, VpT);
  out_gemm_kernel<<<gProj, blk, 0, stream>>>(Ebf, VpT, out);
}
